// LongAttention_53532472377656
// MI455X (gfx1250) — matrix operand order repaired and checked by reading, not compiled, not run on hardware
//
#include <hip/hip_runtime.h>
#include <hip/hip_bf16.h>

// ---------------------------------------------------------------------------
// LongAttention for MI455X (gfx1250, wave32, WMMA).
// B=2 T=4096 C=2048 H=16 D=128 K=4.  NT = B*T = 8192 token rows.
//
// Six 8192x2048x2048 GEMMs run through v_wmma_f32_16x16x32_bf16 (f32 accum)
// with double-buffered LDS tiles filled by global_load_async_to_lds_b128
// (ASYNCcnt-tracked) so HBM traffic overlaps the WMMA math.
// ---------------------------------------------------------------------------

#define NT_TOK 8192
#define T_LEN  4096
#define C_DIM  2048
#define H_NUM  16
#define D_DIM  128
#define K_CONV 4
#define NCHUNK 64          // scan chunks along T
#define CHUNK  (T_LEN / NCHUNK)
#define NCHAN  (2 * H_NUM * D_DIM)   // B*H*D = 4096 scan channels

typedef __bf16 bf16_t;
typedef __attribute__((ext_vector_type(16))) __bf16 v16bf;
typedef __attribute__((ext_vector_type(8)))  float  v8f;
typedef __attribute__((__vector_size__(4 * sizeof(int)))) int vint4;  // native vector

#define LDS_AS __attribute__((address_space(3)))
#define GLB_AS __attribute__((address_space(1)))

__device__ __forceinline__ unsigned short f2bf(float f) {
    unsigned u = __builtin_bit_cast(unsigned, f);
    unsigned r = u + 0x7FFFu + ((u >> 16) & 1u);   // round-to-nearest-even
    return (unsigned short)(r >> 16);
}
__device__ __forceinline__ float sigmoidf(float x) {
    return 1.0f / (1.0f + expf(-x));
}

// 16-byte global -> LDS copy, async (ASYNCcnt) when the toolchain has it.
__device__ __forceinline__ void async_cp16(const unsigned short* g,
                                           unsigned short* l) {
#if __has_builtin(__builtin_amdgcn_global_load_async_to_lds_b128)
    __builtin_amdgcn_global_load_async_to_lds_b128(
        (GLB_AS vint4*)g, (LDS_AS vint4*)l, 0, 0);
#else
    *(vint4*)l = *(const vint4*)g;   // sync fallback
#endif
}
template <int N>
__device__ __forceinline__ void wait_async() {
#if __has_builtin(__builtin_amdgcn_s_wait_asynccnt)
    __builtin_amdgcn_s_wait_asynccnt(N);
#else
    asm volatile("s_wait_asynccnt %0" ::"n"(N));
#endif
}

// ------------------------------- casts -------------------------------------
__global__ void la_cast_bf16_kernel(const float* __restrict__ src,
                                    unsigned short* __restrict__ dst, size_t n) {
    size_t i = (size_t)blockIdx.x * blockDim.x + threadIdx.x;
    if (i < n) dst[i] = f2bf(src[i]);
}

// ------------------- causal depthwise conv (K=4) + SiLU --------------------
__global__ void la_conv_silu_kernel(const float* __restrict__ x,
                                    const float* __restrict__ w,   // (C,1,K)
                                    const float* __restrict__ bias,
                                    float* __restrict__ xc,
                                    unsigned short* __restrict__ xc_bf) {
    size_t idx = (size_t)blockIdx.x * blockDim.x + threadIdx.x;   // over NT*C
    int c = (int)(idx % C_DIM);
    size_t bt = idx / C_DIM;
    int t = (int)(bt % T_LEN);
    int b = (int)(bt / T_LEN);
    float acc = bias[c];
#pragma unroll
    for (int j = 0; j < K_CONV; ++j) {
        int tt = t - (K_CONV - 1) + j;
        if (tt >= 0)
            acc += x[((size_t)b * T_LEN + tt) * C_DIM + c] * w[c * K_CONV + j];
    }
    float s = acc * sigmoidf(acc);   // SiLU
    xc[idx] = s;
    xc_bf[idx] = f2bf(s);
}

// --------------------------- WMMA GEMM: Y = X @ W^T ------------------------
// X: (M,K) bf16 row-major, W: (N,K) bf16 row-major, Y: (M,N) f32.
// Block tile 128x128, BK=32, double-buffered async LDS staging.
// 8 waves/block; wave w computes rows [m0+16w, m0+16w+16) x 128 block cols.
// Epilogues: 0=none, 1=l2norm-over-128, 2=LN-over-128 (g,b), 3=sigmoid(x+bias).
#define EPI_NONE 0
#define EPI_L2   1
#define EPI_LN   2
#define EPI_SIG  3

template <int EPI>
__global__ __launch_bounds__(256)
void la_gemm_bf16_kernel(const unsigned short* __restrict__ A,
                         const unsigned short* __restrict__ W,
                         float* __restrict__ out,
                         const float* __restrict__ bias,   // EPI_SIG: (Ntot)
                         const float* __restrict__ gvec,   // EPI_LN: (128)
                         const float* __restrict__ bvec,   // EPI_LN: (128)
                         int M, int Ntot, int Kdim) {
    __shared__ __align__(32) unsigned short ldsA[2][128 * 32];
    __shared__ __align__(32) unsigned short ldsB[2][128 * 32];

    const int tid  = threadIdx.x;
    const int lane = tid & 31;
    const int wid  = tid >> 5;          // 0..7
    const int ml   = lane & 15;
    const int g    = lane >> 4;         // 0/1 lane group
    const int m0   = blockIdx.x * 128;
    const int n0   = blockIdx.y * 128;

    // cooperative staging: each thread owns 32B of A and 32B of B per tile
    const int ldrow  = tid >> 1;
    const int ldhalf = (tid & 1) * 16;

    v8f acc[8];
    const v8f vzero = {0.f, 0.f, 0.f, 0.f, 0.f, 0.f, 0.f, 0.f};
#pragma unroll
    for (int t = 0; t < 8; ++t) acc[t] = vzero;

    union Frag { vint4 q[2]; v16bf v; };

    auto stage = [&](int buf, int kk) {
        const unsigned short* gA = A + (size_t)(m0 + ldrow) * Kdim + kk + ldhalf;
        const unsigned short* gB = W + (size_t)(n0 + ldrow) * Kdim + kk + ldhalf;
        unsigned short* la = &ldsA[buf][ldrow * 32 + ldhalf];
        unsigned short* lb = &ldsB[buf][ldrow * 32 + ldhalf];
        async_cp16(gA, la);
        async_cp16(gA + 8, la + 8);
        async_cp16(gB, lb);
        async_cp16(gB + 8, lb + 8);
    };

    const int nTiles = Kdim / 32;
    stage(0, 0);
    for (int i = 0; i < nTiles; ++i) {
        const int cur = i & 1;
        if (i + 1 < nTiles) {
            stage(1 - cur, (i + 1) * 32);   // prefetch next tile (async)
            wait_async<4>();                // tile i's 4 copies have landed
        } else {
            wait_async<0>();
        }
        __syncthreads();

        // A fragment: 16x32 bf16, lane m=lane%16, group g holds K halves
        // e0..7 = A[m][8g..8g+7], e8..15 = A[m][16+8g..16+8g+7]
        Frag af;
        const int arow = wid * 16 + ml;
        af.q[0] = *(const vint4*)&ldsA[cur][arow * 32 + 8 * g];
        af.q[1] = *(const vint4*)&ldsA[cur][arow * 32 + 16 + 8 * g];

        // B fragments: 32x16 bf16, lane n=lane%16, same K split as A (8g, 16+8g)
        v16bf bfr[8];
#pragma unroll
        for (int t = 0; t < 8; ++t) {
            Frag bf;
            bf.q[0] = *(const vint4*)&ldsB[cur][(t * 16 + ml) * 32 + 8 * g];
            bf.q[1] = *(const vint4*)&ldsB[cur][(t * 16 + ml) * 32 + 16 + 8 * g];
            bfr[t] = bf.v;
        }

#pragma unroll
        for (int t = 0; t < 8; ++t)
            acc[t] = __builtin_amdgcn_wmma_f32_16x16x32_bf16(
                false, af.v, false, bfr[t], (short)0, acc[t], false, false);

        __syncthreads();   // everyone done reading buf[cur] before reuse
    }

    // ------------------------------ epilogue -------------------------------
    // D layout: acc[t][j] = Y[m0+16*wid + j + 8*g][n0 + 16*t + ml]
#pragma unroll
    for (int j = 0; j < 8; ++j) {
        const int row = m0 + wid * 16 + j + 8 * g;
        if (EPI == EPI_L2 || EPI == EPI_LN) {
            float s = 0.f, ss = 0.f;
#pragma unroll
            for (int t = 0; t < 8; ++t) { float x = acc[t][j]; s += x; ss += x * x; }
            // reduce over the 16 lanes of this group (row has 128 cols)
#pragma unroll
            for (int msk = 8; msk >= 1; msk >>= 1) {
                s  += __shfl_xor(s,  msk, 32);
                ss += __shfl_xor(ss, msk, 32);
            }
            if (EPI == EPI_L2) {
                float scale = 1.0f / fmaxf(sqrtf(ss), 1e-12f);
#pragma unroll
                for (int t = 0; t < 8; ++t)
                    out[(size_t)row * Ntot + n0 + 16 * t + ml] = acc[t][j] * scale;
            } else {
                float mean = s * (1.0f / 128.0f);
                float var  = ss * (1.0f / 128.0f) - mean * mean;
                float inv  = rsqrtf(var + 1e-5f);
#pragma unroll
                for (int t = 0; t < 8; ++t) {
                    int d = 16 * t + ml;
                    float v = (acc[t][j] - mean) * inv * gvec[d] + bvec[d];
                    out[(size_t)row * Ntot + n0 + d] = v;
                }
            }
        } else if (EPI == EPI_SIG) {
#pragma unroll
            for (int t = 0; t < 8; ++t) {
                int n = n0 + 16 * t + ml;
                out[(size_t)row * Ntot + n] = sigmoidf(acc[t][j] + bias[n]);
            }
        } else {
#pragma unroll
            for (int t = 0; t < 8; ++t)
                out[(size_t)row * Ntot + n0 + 16 * t + ml] = acc[t][j];
        }
    }
}

// ------------------- gamma = sigmoid(xc @ gamma_w^T + b) -------------------
// One wave per (token, head): 32-lane strided dot of length C.
__global__ void la_gamma_kernel(const float* __restrict__ xc,
                                const float* __restrict__ gw,   // (H,C)
                                const float* __restrict__ gb,
                                float* __restrict__ gamma) {
    int wg   = (int)((blockIdx.x * blockDim.x + threadIdx.x) >> 5);
    int lane = threadIdx.x & 31;
    int row  = wg >> 4;          // token
    int h    = wg & 15;
    const float* xr = xc + (size_t)row * C_DIM;
    const float* wr = gw + (size_t)h * C_DIM;
    float acc = 0.f;
    for (int c = lane; c < C_DIM; c += 32) acc += xr[c] * wr[c];
#pragma unroll
    for (int msk = 16; msk >= 1; msk >>= 1) acc += __shfl_xor(acc, msk, 32);
    if (lane == 0) gamma[(size_t)row * H_NUM + h] = sigmoidf(acc + gb[h]);
}

// ---------------- chunked decayed scan: m_t = g_t*m_{t-1} + b_t ------------
// channel = (b,h,d); b_t = ig*k*v.  pass1: per-chunk (A=prod g, B=local scan).
__global__ void la_scan_pass1(const float* __restrict__ kb,
                              const float* __restrict__ vb,
                              const float* __restrict__ ib,
                              const float* __restrict__ gamma,
                              float* __restrict__ carA, float* __restrict__ carB) {
    int id = blockIdx.x * blockDim.x + threadIdx.x;   // NCHAN*NCHUNK
    int ch = id >> 6, chunk = id & (NCHUNK - 1);
    int b = ch >> 11, c = ch & (C_DIM - 1), h = c >> 7;
    size_t base = (size_t)b * T_LEN * C_DIM + c;
    size_t gb   = (size_t)b * T_LEN * H_NUM + h;
    float A = 1.f, Bv = 0.f;
    for (int t = chunk * CHUNK; t < chunk * CHUNK + CHUNK; ++t) {
        size_t off = base + (size_t)t * C_DIM;
        float g = gamma[gb + (size_t)t * H_NUM];
        A *= g;
        Bv = g * Bv + ib[off] * kb[off] * vb[off];
    }
    carA[id] = A; carB[id] = Bv;
}

// pass2: per channel, turn chunk aggregates into exclusive carry-ins.
__global__ void la_scan_pass2(float* __restrict__ carA, float* __restrict__ carB) {
    int ch = blockIdx.x * blockDim.x + threadIdx.x;   // NCHAN
    float A = 1.f, Bv = 0.f;
    for (int chunk = 0; chunk < NCHUNK; ++chunk) {
        int idx = ch * NCHUNK + chunk;
        float a = carA[idx], bb = carB[idx];
        carA[idx] = A; carB[idx] = Bv;    // carry entering this chunk
        Bv = a * Bv + bb;
        A  = A * a;
    }
}

// pass3: re-scan each chunk seeded with its carry-in, write mem.
__global__ void la_scan_pass3(const float* __restrict__ kb,
                              const float* __restrict__ vb,
                              const float* __restrict__ ib,
                              const float* __restrict__ gamma,
                              const float* __restrict__ carB,
                              float* __restrict__ mem) {
    int id = blockIdx.x * blockDim.x + threadIdx.x;
    int ch = id >> 6, chunk = id & (NCHUNK - 1);
    int b = ch >> 11, c = ch & (C_DIM - 1), h = c >> 7;
    size_t base = (size_t)b * T_LEN * C_DIM + c;
    size_t gb   = (size_t)b * T_LEN * H_NUM + h;
    float m = carB[id];
    for (int t = chunk * CHUNK; t < chunk * CHUNK + CHUNK; ++t) {
        size_t off = base + (size_t)t * C_DIM;
        float g = gamma[gb + (size_t)t * H_NUM];
        m = g * m + ib[off] * kb[off] * vb[off];
        mem[off] = m;
    }
}

// ----- fused: o = GroupNorm( LN(mem)*q ) * gn + bias, * sigmoid(og) --------
// one wave per (token, head); D=128 reductions via wave32 shuffles.
__global__ void la_epilogue_kernel(const float* __restrict__ mem,
                                   const float* __restrict__ q,
                                   const float* __restrict__ og,
                                   const float* __restrict__ mn_g,
                                   const float* __restrict__ mn_b,
                                   const float* __restrict__ gn_g,
                                   const float* __restrict__ gn_b,
                                   unsigned short* __restrict__ o_bf) {
    int wg   = (int)((blockIdx.x * blockDim.x + threadIdx.x) >> 5);
    int lane = threadIdx.x & 31;
    int token = wg >> 4, h = wg & 15;
    size_t base = (size_t)token * C_DIM + h * D_DIM;

    float xv[4], s = 0.f, ss = 0.f;
#pragma unroll
    for (int i = 0; i < 4; ++i) {
        int d = i * 32 + lane;
        xv[i] = mem[base + d];
        s += xv[i]; ss += xv[i] * xv[i];
    }
#pragma unroll
    for (int msk = 16; msk >= 1; msk >>= 1) { s += __shfl_xor(s, msk, 32); ss += __shfl_xor(ss, msk, 32); }
    float mean = s * (1.0f / 128.0f);
    float var  = ss * (1.0f / 128.0f) - mean * mean;
    float inv  = rsqrtf(var + 1e-5f);

    float ov[4], s2 = 0.f, ss2 = 0.f;
#pragma unroll
    for (int i = 0; i < 4; ++i) {
        int d = i * 32 + lane;
        float t = (xv[i] - mean) * inv * mn_g[d] + mn_b[d];
        t *= q[base + d];
        ov[i] = t; s2 += t; ss2 += t * t;
    }
#pragma unroll
    for (int msk = 16; msk >= 1; msk >>= 1) { s2 += __shfl_xor(s2, msk, 32); ss2 += __shfl_xor(ss2, msk, 32); }
    float m2  = s2 * (1.0f / 128.0f);
    float v2  = ss2 * (1.0f / 128.0f) - m2 * m2;
    float iv2 = rsqrtf(v2 + 1e-5f);

#pragma unroll
    for (int i = 0; i < 4; ++i) {
        int d = i * 32 + lane;
        int c = h * D_DIM + d;
        float o = (ov[i] - m2) * iv2 * gn_g[c] + gn_b[c];
        o *= og[(size_t)token * C_DIM + c];
        o_bf[(size_t)token * C_DIM + c] = f2bf(o);
    }
}

// ---------------------------------------------------------------------------
extern "C" void kernel_launch(void* const* d_in, const int* in_sizes, int n_in,
                              void* d_out, int out_size, void* d_ws, size_t ws_size,
                              hipStream_t stream) {
    const float* x       = (const float*)d_in[0];
    const float* Wq      = (const float*)d_in[1];
    const float* Wk      = (const float*)d_in[2];
    const float* Wv      = (const float*)d_in[3];
    const float* Wo      = (const float*)d_in[4];
    const float* conv_w  = (const float*)d_in[5];
    const float* conv_b  = (const float*)d_in[6];
    const float* ig_w    = (const float*)d_in[7];
    const float* ig_b    = (const float*)d_in[8];
    const float* og_w    = (const float*)d_in[9];
    const float* og_b    = (const float*)d_in[10];
    const float* gamma_w = (const float*)d_in[11];
    const float* gamma_b = (const float*)d_in[12];
    const float* vn_g    = (const float*)d_in[13];
    const float* vn_b    = (const float*)d_in[14];
    const float* gn_g    = (const float*)d_in[15];
    const float* gn_b    = (const float*)d_in[16];
    const float* mn_g    = (const float*)d_in[17];
    const float* mn_b    = (const float*)d_in[18];
    float* outp = (float*)d_out;          // also used as q (same NT*C f32 size)

    const size_t NC = (size_t)NT_TOK * C_DIM;
    const size_t CC = (size_t)C_DIM * C_DIM;

    char* p = (char*)d_ws;
    auto alloc = [&](size_t bytes) -> void* {
        void* r = (void*)p;
        p += (bytes + 255) & ~(size_t)255;
        return r;
    };
    float*          xc_f   = (float*)alloc(NC * 4);
    unsigned short* xc_bf  = (unsigned short*)alloc(NC * 2);
    unsigned short* x_bf   = (unsigned short*)alloc(NC * 2);
    unsigned short* wq_bf  = (unsigned short*)alloc(CC * 2);
    unsigned short* wk_bf  = (unsigned short*)alloc(CC * 2);
    unsigned short* wv_bf  = (unsigned short*)alloc(CC * 2);
    unsigned short* wig_bf = (unsigned short*)alloc(CC * 2);
    unsigned short* wog_bf = (unsigned short*)alloc(CC * 2);
    unsigned short* wo_bf  = (unsigned short*)alloc(CC * 2);
    float*          kbuf   = (float*)alloc(NC * 4);
    float*          vbuf   = (float*)alloc(NC * 4);
    float*          igbuf  = (float*)alloc(NC * 4);
    float*          ogbuf  = (float*)alloc(NC * 4);
    float*          gammab = (float*)alloc((size_t)NT_TOK * H_NUM * 4);
    float*          membuf = (float*)alloc(NC * 4);
    unsigned short* o_bf   = (unsigned short*)alloc(NC * 2);
    float*          carA   = (float*)alloc((size_t)NCHAN * NCHUNK * 4);
    float*          carB   = (float*)alloc((size_t)NCHAN * NCHUNK * 4);
    (void)ws_size; (void)n_in; (void)in_sizes; (void)out_size;

    // 1) casts + conv/SiLU
    la_cast_bf16_kernel<<<NC / 256, 256, 0, stream>>>(x, x_bf, NC);
    la_cast_bf16_kernel<<<CC / 256, 256, 0, stream>>>(Wq, wq_bf, CC);
    la_cast_bf16_kernel<<<CC / 256, 256, 0, stream>>>(Wk, wk_bf, CC);
    la_cast_bf16_kernel<<<CC / 256, 256, 0, stream>>>(Wv, wv_bf, CC);
    la_cast_bf16_kernel<<<CC / 256, 256, 0, stream>>>(ig_w, wig_bf, CC);
    la_cast_bf16_kernel<<<CC / 256, 256, 0, stream>>>(og_w, wog_bf, CC);
    la_cast_bf16_kernel<<<CC / 256, 256, 0, stream>>>(Wo, wo_bf, CC);
    la_conv_silu_kernel<<<NC / 256, 256, 0, stream>>>(x, conv_w, conv_b, xc_f, xc_bf);

    // 2) projection GEMMs (fused norms / gates)
    dim3 gg(NT_TOK / 128, C_DIM / 128);
    la_gemm_bf16_kernel<EPI_L2><<<gg, 256, 0, stream>>>(
        x_bf, wq_bf, outp, nullptr, nullptr, nullptr, NT_TOK, C_DIM, C_DIM); // q -> d_out
    la_gemm_bf16_kernel<EPI_L2><<<gg, 256, 0, stream>>>(
        x_bf, wk_bf, kbuf, nullptr, nullptr, nullptr, NT_TOK, C_DIM, C_DIM);
    la_gemm_bf16_kernel<EPI_LN><<<gg, 256, 0, stream>>>(
        x_bf, wv_bf, vbuf, nullptr, vn_g, vn_b, NT_TOK, C_DIM, C_DIM);
    la_gemm_bf16_kernel<EPI_SIG><<<gg, 256, 0, stream>>>(
        xc_bf, wig_bf, igbuf, ig_b, nullptr, nullptr, NT_TOK, C_DIM, C_DIM);
    la_gemm_bf16_kernel<EPI_SIG><<<gg, 256, 0, stream>>>(
        xc_bf, wog_bf, ogbuf, og_b, nullptr, nullptr, NT_TOK, C_DIM, C_DIM);

    // 3) gamma + chunked scan
    la_gamma_kernel<<<(NT_TOK * H_NUM) / 8, 256, 0, stream>>>(xc_f, gamma_w, gamma_b, gammab);
    la_scan_pass1<<<(NCHAN * NCHUNK) / 256, 256, 0, stream>>>(kbuf, vbuf, igbuf, gammab, carA, carB);
    la_scan_pass2<<<NCHAN / 256, 256, 0, stream>>>(carA, carB);
    la_scan_pass3<<<(NCHAN * NCHUNK) / 256, 256, 0, stream>>>(kbuf, vbuf, igbuf, gammab, carB, membuf);

    // 4) fused LN*q -> GroupNorm -> *sigmoid(og), emit bf16
    la_epilogue_kernel<<<(NT_TOK * H_NUM) / 8, 256, 0, stream>>>(
        membuf, outp, ogbuf, mn_g, mn_b, gn_g, gn_b, o_bf);

    // 5) final projection: d_out = o @ Wo^T
    la_gemm_bf16_kernel<EPI_NONE><<<gg, 256, 0, stream>>>(
        o_bf, wo_bf, outp, nullptr, nullptr, nullptr, NT_TOK, C_DIM, C_DIM);
}
